// MultiheadAttention_27547920236839
// MI455X (gfx1250) — matrix operand order repaired and checked by reading, not compiled, not run on hardware
//
#include <hip/hip_runtime.h>

// ---------------------------------------------------------------- constants
#define TGT   2048
#define BSZ   2
#define EMB   1024
#define NHEAD 16          // heads per batch
#define HEADS 32          // BSZ*NHEAD
#define HD    64          // head dim
#define MROWS (TGT*BSZ)   // 4096 rows (t,b)
#define NQKV  (3*EMB)     // 3072

typedef __bf16 bf16_t;
typedef bf16_t v16bf __attribute__((ext_vector_type(16)));
typedef float  v8f   __attribute__((ext_vector_type(8)));

__device__ inline v8f v8f_zero() {
    v8f z = {0.f, 0.f, 0.f, 0.f, 0.f, 0.f, 0.f, 0.f};
    return z;
}

// 16-bit A-matrix 16x32 fragment (M=row, K contiguous in memory).
// lanes 0-15: row m0+l, K = k0+{0..7,16..23}; lanes 16-31: row m0+(l-16), K = k0+{8..15,24..31}
__device__ inline v16bf load_frag_a(const bf16_t* __restrict__ base, int lda,
                                    int m0, int k0, int lane) {
    const int r  = lane & 15;
    const int hi = lane >> 4;
    const bf16_t* p = base + (size_t)(m0 + r) * lda + k0;
    union { v16bf v; int4 q[2]; } u;
    u.q[0] = *(const int4*)(p + (hi ? 8 : 0));
    u.q[1] = *(const int4*)(p + (hi ? 24 : 16));
    return u.v;
}

// 16-bit B-matrix 32x16 fragment. B(k,n) with memory holding "row n" K-contiguous
// (NT gemm). lanes 0-15: N=n0+l, K=k0+{0..7,16..23}; lanes 16-31: N=n0+(l-16), K=k0+{8..15,24..31}
__device__ inline v16bf load_frag_b(const bf16_t* __restrict__ base, int ldb,
                                    int n0, int k0, int lane) {
    const int c  = lane & 15;
    const int hi = lane >> 4;
    const bf16_t* p = base + (size_t)(n0 + c) * ldb + k0 + (hi ? 8 : 0);
    union { v16bf v; int4 q[2]; } u;
    u.q[0] = *(const int4*)(p);
    u.q[1] = *(const int4*)(p + 16);
    return u.v;
}

#define WMMA_BF16(A, B, C) \
    __builtin_amdgcn_wmma_f32_16x16x32_bf16(false, (A), false, (B), (short)0, (C), false, false)

// ---------------------------------------------------------------- fp32 -> bf16
__global__ void cvt_f32_to_bf16(const float* __restrict__ src,
                                bf16_t* __restrict__ dst, int n) {
    int i = blockIdx.x * blockDim.x + threadIdx.x;
    if (i < n) dst[i] = (bf16_t)src[i];
}

// ---------------------------------------------------------------- QKV projection
// C[m,n] = sum_k A[m,k] * W[n,k] + bias[n]; scatter into packed per-head bf16:
//   q -> qh[head][t][d], k -> kh[head][t][d], v -> vT[head][d][t] (pre-transposed)
__global__ __launch_bounds__(256)
void gemm_qkv(const bf16_t* __restrict__ A, const bf16_t* __restrict__ W,
              const float* __restrict__ bias,
              bf16_t* __restrict__ qh, bf16_t* __restrict__ kh,
              bf16_t* __restrict__ vT) {
    const int lane = threadIdx.x & 31;
    const int wave = threadIdx.x >> 5;
    const int m0 = blockIdx.y * 128 + (wave >> 2) * 64;
    const int n0 = blockIdx.x * 128 + (wave & 3) * 32;
    const int rbase = (lane >> 4) * 8;
    const int cl    = lane & 15;

    v8f acc[4][2];
#pragma unroll
    for (int i = 0; i < 4; ++i)
#pragma unroll
        for (int j = 0; j < 2; ++j) acc[i][j] = v8f_zero();

    for (int k = 0; k < EMB; k += 32) {
        if (k + 32 < EMB) {
            __builtin_prefetch((const void*)(A + (size_t)(m0 + cl) * EMB + k + 32), 0, 0);
            __builtin_prefetch((const void*)(W + (size_t)(n0 + cl) * EMB + k + 32), 0, 0);
        }
        v16bf b0 = load_frag_b(W, EMB, n0,      k, lane);
        v16bf b1 = load_frag_b(W, EMB, n0 + 16, k, lane);
#pragma unroll
        for (int mi = 0; mi < 4; ++mi) {
            v16bf af = load_frag_a(A, EMB, m0 + mi * 16, k, lane);
            acc[mi][0] = WMMA_BF16(af, b0, acc[mi][0]);
            acc[mi][1] = WMMA_BF16(af, b1, acc[mi][1]);
        }
    }

#pragma unroll
    for (int mi = 0; mi < 4; ++mi)
#pragma unroll
        for (int ni = 0; ni < 2; ++ni)
#pragma unroll
            for (int r = 0; r < 8; ++r) {
                const int m = m0 + mi * 16 + rbase + r;
                const int n = n0 + ni * 16 + cl;
                const float val = acc[mi][ni][r] + bias[n];
                const int sec = n >> 10;           // 0=q 1=k 2=v
                const int e2 = n & 1023;
                const int h = e2 >> 6, d = e2 & 63;
                const int b = m & 1, t = m >> 1;
                const int head = (b << 4) | h;
                if (sec == 0)
                    qh[((size_t)head * TGT + t) * HD + d] = (bf16_t)val;
                else if (sec == 1)
                    kh[((size_t)head * TGT + t) * HD + d] = (bf16_t)val;
                else
                    vT[((size_t)head * HD + d) * TGT + t] = (bf16_t)val;
            }
}

// ---------------------------------------------------------------- fused attention
// One block = 4 waves = 16 query rows of one head. Scores 16x2048 stay in LDS.
// QK^T via WMMA -> softmax in LDS (weights written once to d_out) -> P*V via WMMA.
__global__ __launch_bounds__(128)
void attn_fused(const bf16_t* __restrict__ qh, const bf16_t* __restrict__ kh,
                const bf16_t* __restrict__ vT, float* __restrict__ wts,
                bf16_t* __restrict__ ctx) {
    extern __shared__ float smem[];
    float* scores = smem;                 // 16 * 2048
    float* redm   = smem + 16 * TGT;      // 128
    float* reds   = redm + 128;           // 128

    const int lane = threadIdx.x & 31;
    const int wave = threadIdx.x >> 5;
    const int head = blockIdx.y;
    const int t0   = blockIdx.x * 16;
    const int rbase = (lane >> 4) * 8;
    const int cl    = lane & 15;

    const bf16_t* qhead  = qh + (size_t)head * TGT * HD;
    const bf16_t* khead  = kh + (size_t)head * TGT * HD;
    const bf16_t* vthead = vT + (size_t)head * HD * TGT;

    // ---- Stage 1: scores = (Q K^T) * rsqrt(64); wave owns a 512-col strip
    {
        v16bf a0 = load_frag_a(qhead, HD, t0, 0, lane);
        v16bf a1 = load_frag_a(qhead, HD, t0, 32, lane);
        for (int s0 = wave * 512; s0 < wave * 512 + 512; s0 += 16) {
            v16bf b0 = load_frag_b(khead, HD, s0, 0, lane);
            v16bf b1 = load_frag_b(khead, HD, s0, 32, lane);
            v8f acc = v8f_zero();
            acc = WMMA_BF16(a0, b0, acc);
            acc = WMMA_BF16(a1, b1, acc);
#pragma unroll
            for (int r = 0; r < 8; ++r)
                scores[(size_t)(rbase + r) * TGT + s0 + cl] = acc[r] * 0.125f;
        }
    }
    __syncthreads();

    // ---- Stage 2: softmax over S in LDS; write normalized weights to global
    {
        const int row = threadIdx.x >> 3;   // 16 rows, 8 threads each
        const int seg = threadIdx.x & 7;
        float* srow = scores + (size_t)row * TGT;
        float mx = -3.402823466e+38f;
        for (int j = seg; j < TGT; j += 8) mx = fmaxf(mx, srow[j]);
        redm[row * 8 + seg] = mx;
        __syncthreads();
        float m = redm[row * 8];
#pragma unroll
        for (int i = 1; i < 8; ++i) m = fmaxf(m, redm[row * 8 + i]);
        float sum = 0.f;
        for (int j = seg; j < TGT; j += 8) {
            float e = __expf(srow[j] - m);
            srow[j] = e;
            sum += e;
        }
        reds[row * 8 + seg] = sum;
        __syncthreads();
        float tot = 0.f;
#pragma unroll
        for (int i = 0; i < 8; ++i) tot += reds[row * 8 + i];
        const float inv = 1.f / tot;
        float* wrow = wts + ((size_t)head * TGT + t0 + row) * TGT;
        for (int j = seg; j < TGT; j += 8) {
            float w = srow[j] * inv;
            srow[j] = w;       // normalized weights stay in LDS for P*V
            wrow[j] = w;       // single global write of the weights output
        }
    }
    __syncthreads();

    // ---- Stage 3: ctx(16x64) = W(16x2048) * V(2048x64); wave owns 16 d-cols
    {
        const int d0 = wave * 16;
        v8f acc = v8f_zero();
        const float* prow = scores + (size_t)cl * TGT;
        const int o1 = (lane >> 4) ? 8 : 0;
        const int o2 = (lane >> 4) ? 24 : 16;
        for (int k = 0; k < TGT; k += 32) {
            const float* pr = prow + k;
            float4 f0 = *(const float4*)(pr + o1);
            float4 f1 = *(const float4*)(pr + o1 + 4);
            float4 f2 = *(const float4*)(pr + o2);
            float4 f3 = *(const float4*)(pr + o2 + 4);
            v16bf a;
            a[0] = (bf16_t)f0.x;  a[1] = (bf16_t)f0.y;  a[2] = (bf16_t)f0.z;  a[3] = (bf16_t)f0.w;
            a[4] = (bf16_t)f1.x;  a[5] = (bf16_t)f1.y;  a[6] = (bf16_t)f1.z;  a[7] = (bf16_t)f1.w;
            a[8] = (bf16_t)f2.x;  a[9] = (bf16_t)f2.y;  a[10] = (bf16_t)f2.z; a[11] = (bf16_t)f2.w;
            a[12] = (bf16_t)f3.x; a[13] = (bf16_t)f3.y; a[14] = (bf16_t)f3.z; a[15] = (bf16_t)f3.w;
            v16bf b = load_frag_b(vthead, TGT, d0, k, lane);  // V^T: K-contiguous
            acc = WMMA_BF16(a, b, acc);
        }
        const int bq = head >> 4, h = head & 15;
#pragma unroll
        for (int r = 0; r < 8; ++r) {
            const int t = t0 + rbase + r;
            const int d = d0 + cl;
            ctx[((size_t)t * BSZ + bq) * EMB + h * HD + d] = (bf16_t)acc[r];
        }
    }
}

// ---------------------------------------------------------------- output projection
// out[m,n] = sum_k ctx[m,k] * Wout[n,k] + bias[n]   (fp32 store into d_out)
__global__ __launch_bounds__(256)
void gemm_out(const bf16_t* __restrict__ A, const bf16_t* __restrict__ W,
              const float* __restrict__ bias, float* __restrict__ out) {
    const int lane = threadIdx.x & 31;
    const int wave = threadIdx.x >> 5;
    const int m0 = blockIdx.y * 128 + (wave >> 2) * 64;
    const int n0 = blockIdx.x * 128 + (wave & 3) * 32;
    const int rbase = (lane >> 4) * 8;
    const int cl    = lane & 15;

    v8f acc[4][2];
#pragma unroll
    for (int i = 0; i < 4; ++i)
#pragma unroll
        for (int j = 0; j < 2; ++j) acc[i][j] = v8f_zero();

    for (int k = 0; k < EMB; k += 32) {
        if (k + 32 < EMB) {
            __builtin_prefetch((const void*)(A + (size_t)(m0 + cl) * EMB + k + 32), 0, 0);
            __builtin_prefetch((const void*)(W + (size_t)(n0 + cl) * EMB + k + 32), 0, 0);
        }
        v16bf b0 = load_frag_b(W, EMB, n0,      k, lane);
        v16bf b1 = load_frag_b(W, EMB, n0 + 16, k, lane);
#pragma unroll
        for (int mi = 0; mi < 4; ++mi) {
            v16bf af = load_frag_a(A, EMB, m0 + mi * 16, k, lane);
            acc[mi][0] = WMMA_BF16(af, b0, acc[mi][0]);
            acc[mi][1] = WMMA_BF16(af, b1, acc[mi][1]);
        }
    }

#pragma unroll
    for (int mi = 0; mi < 4; ++mi)
#pragma unroll
        for (int ni = 0; ni < 2; ++ni)
#pragma unroll
            for (int r = 0; r < 8; ++r) {
                const int m = m0 + mi * 16 + rbase + r;
                const int n = n0 + ni * 16 + cl;
                out[(size_t)m * EMB + n] = acc[mi][ni][r] + bias[n];
            }
}

// ---------------------------------------------------------------- launcher
extern "C" void kernel_launch(void* const* d_in, const int* in_sizes, int n_in,
                              void* d_out, int out_size, void* d_ws, size_t ws_size,
                              hipStream_t stream) {
    (void)in_sizes; (void)n_in; (void)out_size; (void)ws_size;

    const float* query = (const float*)d_in[0];
    // d_in[1] (key) and d_in[2] (value) are unused: the reference derives q,k,v
    // all from `query` (bug kept faithfully).
    const float* w_in  = (const float*)d_in[3];
    const float* b_in  = (const float*)d_in[4];
    const float* w_out = (const float*)d_in[5];
    const float* b_out = (const float*)d_in[6];

    char* ws = (char*)d_ws;
    bf16_t* A_bf  = (bf16_t*)(ws);                 // 4096*1024  (8 MB)
    bf16_t* W1_bf = (bf16_t*)(ws + 8388608);       // 3072*1024  (6 MB)
    bf16_t* W2_bf = (bf16_t*)(ws + 14680064);      // 1024*1024  (2 MB)
    bf16_t* qh    = (bf16_t*)(ws + 16777216);      // 32*2048*64 (8 MB)
    bf16_t* kh    = (bf16_t*)(ws + 25165824);      // 8 MB
    bf16_t* vT    = (bf16_t*)(ws + 33554432);      // 8 MB  (per-head transposed V)
    bf16_t* ctx   = (bf16_t*)(ws + 41943040);      // 4096*1024 (8 MB)

    float* out_attn = (float*)d_out;                                 // 4096*1024
    float* out_wts  = out_attn + (size_t)MROWS * EMB;                // 32*2048*2048

    cvt_f32_to_bf16<<<(MROWS * EMB) / 256, 256, 0, stream>>>(query, A_bf, MROWS * EMB);
    cvt_f32_to_bf16<<<(NQKV * EMB) / 256, 256, 0, stream>>>(w_in, W1_bf, NQKV * EMB);
    cvt_f32_to_bf16<<<(EMB * EMB) / 256, 256, 0, stream>>>(w_out, W2_bf, EMB * EMB);

    gemm_qkv<<<dim3(NQKV / 128, MROWS / 128), 256, 0, stream>>>(
        A_bf, W1_bf, b_in, qh, kh, vT);

    const size_t attn_lds = (size_t)(16 * TGT + 256) * sizeof(float); // 132 KB
    attn_fused<<<dim3(TGT / 16, HEADS), 128, attn_lds, stream>>>(
        qh, kh, vT, out_wts, ctx);

    gemm_out<<<dim3(EMB / 128, MROWS / 128), 256, 0, stream>>>(
        ctx, W2_bf, b_out, out_attn);
}
